// SpectralBandProcessor_26912265076907
// MI455X (gfx1250) — matrix operand order repaired and checked by reading, not compiled, not run on hardware
//
#include <hip/hip_runtime.h>
#include <math.h>

// ---------------------------------------------------------------------------
// SpectralBandProcessor fused kernels for gfx1250 (MI455X)
//   K1: per (band,batch,16x16-tile):
//       conv1(3x3,1->64)+gelu -> LDS f16 (transposed [pos][ic] for b128 B-frags)
//       conv2(3x3,64->64) as 9 per-tap WMMA f16 GEMMs (K=64 each),
//         both 8-row halves of the tile per k-step (A fragment loaded once,
//         consumed twice -> no LICM hoist/spill of A fragments)
//       conv3(1x1,64->1) reduced in-register -> processed map (d_ws)
//   K2: cross-band 1x1 fusion 31->62->31 + residual
// ---------------------------------------------------------------------------

#define NB     31
#define PD     64
#define HW     192
#define TS     16
#define TILES  12        // 192 / 16
#define H1DIM  18        // TS + 2 (halo 1 for conv2)
#define H1N    324       // 18*18
#define XDIM   20        // TS + 4
#define ICSTR  72        // h1T row stride in halfs: 36 dwords (conflict-free), 16B aligned
#define W2SZ   (9*2*4*32*16)   // 36864 halfs, fragment-swizzled
#define ROWOFF (8 * H1DIM * ICSTR)   // h1T offset between row y and y+8

typedef __attribute__((ext_vector_type(16))) _Float16     v16h;
typedef __attribute__((ext_vector_type(8)))  float        v8f;
typedef __attribute__((ext_vector_type(4)))  unsigned int u32x4;

union FRAG16 { v16h h; u32x4 q[2]; };

// Branch-free GELU: x * sigmoid(1.5957692*(x + 0.044715 x^3))
// (tanh-form GELU; hardware v_exp_f32 / v_rcp_f32, no EXEC divergence)
__device__ __forceinline__ float gelu_fast(float v) {
    float u = v * (v * v * 0.044715f + 1.0f);
    float e = __expf(-1.5957691216057308f * u);
    return v * __builtin_amdgcn_rcpf(1.0f + e);
}

// ---------------------------------------------------------------------------
// Kernel 1: fused per-band 3-layer stack
// grid = (144, 31, 2), block = 256 (8 waves of 32)
// ---------------------------------------------------------------------------
__global__ __launch_bounds__(256) void band_kernel(
    const float* __restrict__ x,  const float* __restrict__ W1,
    const float* __restrict__ b1, const float* __restrict__ W2,
    const float* __restrict__ b2, const float* __restrict__ W3,
    const float* __restrict__ b3, float* __restrict__ proc)
{
    // A-matrix fragments, pre-swizzled: [(tap*2+ks2)*4+mt][lane][e]  (73,728 B)
    __shared__ __align__(16) _Float16 ldsW2[W2SZ];
    // conv1 output, transposed [pos][ic], padded row               (46,656 B)
    __shared__ __align__(16) _Float16 ldsH1T[H1N * ICSTR];
    __shared__ __align__(16) float    ldsX[XDIM * XDIM];            //  1,600 B
    __shared__ __align__(16) float    ldsW1[PD * 9];                //  2,304 B
    __shared__ float ldsB1[PD], ldsB2[PD], ldsW3[PD];

    const int tid   = threadIdx.x;
    const int ty    = blockIdx.x / TILES;
    const int tx    = blockIdx.x % TILES;
    const int band  = blockIdx.y;
    const int batch = blockIdx.z;

    // ---------------- stage 0: stage weights / input tile into LDS ----------
    // W2 band slice: [oc][ic*9 + tap], staged directly into WMMA-A fragment order:
    //   linear i = (((tap*2+ks2)*4+mt)*32+lane)*16+e
    //   lane: hi=lane>>4, m=lane&15 ; oc = mt*16+m
    //   ic   = ks2*32 + e + (hi? 16 : 0)      (contiguous k = 16*hi + e, same order as B)
    const float* W2g = W2 + (size_t)band * PD * 576;
    for (int i = tid; i < W2SZ; i += 256) {
        int e   = i & 15;
        int ln  = (i >> 4) & 31;
        int mt  = (i >> 9) & 3;
        int ks2 = (i >> 11) & 1;
        int tap = i >> 12;
        int oc  = mt * 16 + (ln & 15);
        int ic  = ks2 * 32 + e + ((ln >> 4) ? 16 : 0);
        ldsW2[i] = (_Float16)W2g[oc * 576 + ic * 9 + tap];
    }
    for (int i = tid; i < PD * 9; i += 256) ldsW1[i] = W1[band * PD * 9 + i];
    for (int i = tid; i < PD; i += 256) {
        ldsB1[i] = b1[band * PD + i];
        ldsB2[i] = b2[band * PD + i];
        ldsW3[i] = W3[band * PD + i];
    }
    const float* xb = x + ((size_t)batch * NB + band) * (HW * HW);
    for (int i = tid; i < XDIM * XDIM; i += 256) {
        int r = i / XDIM, c = i - r * XDIM;
        int gy = ty * TS + r - 2, gx = tx * TS + c - 2;
        float v = 0.f;
        if ((unsigned)gy < HW && (unsigned)gx < HW) v = xb[gy * HW + gx];
        ldsX[i] = v;
    }
    __syncthreads();

    // ---------------- stage 1: h1 = gelu(conv1) on 18x18, transposed --------
    for (int p = tid; p < H1N; p += 256) {
        int py = p / H1DIM, px = p - py * H1DIM;
        float t0 = ldsX[(py+0)*XDIM + px+0], t1 = ldsX[(py+0)*XDIM + px+1], t2 = ldsX[(py+0)*XDIM + px+2];
        float t3 = ldsX[(py+1)*XDIM + px+0], t4 = ldsX[(py+1)*XDIM + px+1], t5 = ldsX[(py+1)*XDIM + px+2];
        float t6 = ldsX[(py+2)*XDIM + px+0], t7 = ldsX[(py+2)*XDIM + px+1], t8 = ldsX[(py+2)*XDIM + px+2];
        // conv2 zero-pads its *input*; h1 outside the image must be exactly 0
        float inimg = (((unsigned)(ty*TS + py - 1) < HW) && ((unsigned)(tx*TS + px - 1) < HW)) ? 1.f : 0.f;
        _Float16* hrow = &ldsH1T[p * ICSTR];
        #pragma unroll 8
        for (int c = 0; c < PD; ++c) {
            const float* w = &ldsW1[c * 9];
            float a = ldsB1[c]
                    + t0*w[0] + t1*w[1] + t2*w[2]
                    + t3*w[3] + t4*w[4] + t5*w[5]
                    + t6*w[6] + t7*w[7] + t8*w[8];
            hrow[c] = (_Float16)(inimg * gelu_fast(a));
        }
    }
    __syncthreads();

    // ---------------- stage 2: conv2 as 9 tap-GEMMs via WMMA + conv3 --------
    const int lane = tid & 31;
    const int wv   = tid >> 5;          // wave id 0..7
    const int hi   = lane >> 4;         // half-wave
    const int n    = lane & 15;         // N = pixel column within 16-wide row
    const int hiB  = hi ? 16 : 0;       // B-fragment K offset (halfs)
    const float b3v = b3[band];

    const int y0 = wv;                  // rows y0 and y0+8 handled together
    // acc[0..3]: rows y0, mt 0..3 ; acc[4..7]: rows y0+8
    v8f acc[8] = {v8f{}, v8f{}, v8f{}, v8f{}, v8f{}, v8f{}, v8f{}, v8f{}};

    #pragma unroll 1                    // keep tap as a real loop: bounds live range,
    for (int tap = 0; tap < 9; ++tap) { // prevents whole-loop A-fragment hoist/spill
        const int ky = tap / 3, kx = tap - (tap / 3) * 3;
        const int pos0 = ((y0 + ky) * H1DIM + (n + kx)) * ICSTR;
        #pragma unroll
        for (int ks2 = 0; ks2 < 2; ++ks2) {
            // B fragments for both row-halves: 2x (2x ds_load_b128)
            FRAG16 bf0, bf1;
            const _Float16* bp = &ldsH1T[pos0 + ks2 * 32 + hiB];
            bf0.q[0] = *reinterpret_cast<const u32x4*>(bp);
            bf0.q[1] = *reinterpret_cast<const u32x4*>(bp + 8);
            bf1.q[0] = *reinterpret_cast<const u32x4*>(bp + ROWOFF);
            bf1.q[1] = *reinterpret_cast<const u32x4*>(bp + ROWOFF + 8);
            #pragma unroll
            for (int mt = 0; mt < 4; ++mt) {
                // A fragment: pre-swizzled, 32 contiguous bytes, loaded once,
                // consumed immediately by both row-halves
                FRAG16 af;
                const _Float16* ap =
                    &ldsW2[((((tap * 2 + ks2) * 4 + mt) * 32) + lane) * 16];
                af.q[0] = *reinterpret_cast<const u32x4*>(ap);
                af.q[1] = *reinterpret_cast<const u32x4*>(ap + 8);
                acc[mt]     = __builtin_amdgcn_wmma_f32_16x16x32_f16(
                    false, af.h, false, bf0.h, (short)0, acc[mt],     false, false);
                acc[mt + 4] = __builtin_amdgcn_wmma_f32_16x16x32_f16(
                    false, af.h, false, bf1.h, (short)0, acc[mt + 4], false, false);
            }
        }
    }

    // epilogue: + b2, gelu, x W3, reduce over 64 channels, + b3 -> processed
    float part0 = 0.f, part1 = 0.f;
    #pragma unroll
    for (int mt = 0; mt < 4; ++mt) {
        #pragma unroll
        for (int j = 0; j < 8; ++j) {
            int ch = mt * 16 + j + (hi ? 8 : 0);     // C layout: VGPR j -> M=j (+8 hi)
            float w3 = ldsW3[ch], bb = ldsB2[ch];
            part0 += gelu_fast(acc[mt][j]     + bb) * w3;
            part1 += gelu_fast(acc[mt + 4][j] + bb) * w3;
        }
    }
    part0 += __shfl_xor(part0, 16, 32);              // combine the two half-waves
    part1 += __shfl_xor(part1, 16, 32);
    if (hi == 0) {
        const size_t rowBase = (((size_t)batch * NB + band) * HW + (ty * TS)) * HW
                             + (size_t)(tx * TS) + n;
        proc[rowBase + (size_t)y0 * HW]       = part0 + b3v;
        proc[rowBase + (size_t)(y0 + 8) * HW] = part1 + b3v;
    }
}

// ---------------------------------------------------------------------------
// Kernel 2: cross-band fusion 31 -> 62 -> 31 (1x1 convs) + residual
// one thread per pixel; grid = 2*192*192/256 = 288 blocks
// ---------------------------------------------------------------------------
__global__ __launch_bounds__(256) void fuse_kernel(
    const float* __restrict__ x,   const float* __restrict__ Wf1,
    const float* __restrict__ bf1, const float* __restrict__ Wf2,
    const float* __restrict__ bf2, const float* __restrict__ proc,
    float* __restrict__ out)
{
    __shared__ float sW1[2 * NB * NB], sB1[2 * NB], sW2[NB * 2 * NB], sB2[NB];
    const int tid = threadIdx.x;
    for (int i = tid; i < 2 * NB * NB; i += 256) { sW1[i] = Wf1[i]; sW2[i] = Wf2[i]; }
    for (int i = tid; i < 2 * NB; i += 256) sB1[i] = bf1[i];
    for (int i = tid; i < NB;     i += 256) sB2[i] = bf2[i];
    __syncthreads();

    const int gid   = blockIdx.x * 256 + tid;            // 0 .. 73727 (exact)
    const int batch = gid / (HW * HW);
    const int pix   = gid - batch * (HW * HW);
    const size_t base = (size_t)batch * NB * HW * HW + pix;

    float p[NB];
    #pragma unroll
    for (int c = 0; c < NB; ++c) p[c] = proc[base + (size_t)c * HW * HW];

    float o[NB];
    #pragma unroll
    for (int c = 0; c < NB; ++c) o[c] = sB2[c];

    for (int j = 0; j < 2 * NB; ++j) {
        float f = sB1[j];
        #pragma unroll
        for (int c = 0; c < NB; ++c) f += sW1[j * NB + c] * p[c];
        f = gelu_fast(f);
        #pragma unroll
        for (int c = 0; c < NB; ++c) o[c] += sW2[c * 2 * NB + j] * f;
    }
    #pragma unroll
    for (int c = 0; c < NB; ++c)
        out[base + (size_t)c * HW * HW] = x[base + (size_t)c * HW * HW] + o[c];
}

// ---------------------------------------------------------------------------
extern "C" void kernel_launch(void* const* d_in, const int* in_sizes, int n_in,
                              void* d_out, int out_size, void* d_ws, size_t ws_size,
                              hipStream_t stream) {
    const float* x   = (const float*)d_in[0];
    const float* W1  = (const float*)d_in[1];
    const float* b1  = (const float*)d_in[2];
    const float* W2  = (const float*)d_in[3];
    const float* b2  = (const float*)d_in[4];
    const float* W3  = (const float*)d_in[5];
    const float* b3  = (const float*)d_in[6];
    const float* Wf1 = (const float*)d_in[7];
    const float* bf1 = (const float*)d_in[8];
    const float* Wf2 = (const float*)d_in[9];
    const float* bf2 = (const float*)d_in[10];

    float* proc = (float*)d_ws;            // 2*31*192*192 fp32 = 9.14 MB scratch
    float* out  = (float*)d_out;

    dim3 grid1(TILES * TILES, NB, 2);
    band_kernel<<<grid1, 256, 0, stream>>>(x, W1, b1, W2, b2, W3, b3, proc);

    fuse_kernel<<<(2 * HW * HW) / 256, 256, 0, stream>>>(x, Wf1, bf1, Wf2, bf2, proc, out);
}
